// Seq2SeqWithAttention_52235392254479
// MI455X (gfx1250) — matrix operand order repaired and checked by reading, not compiled, not run on hardware
//
#include <hip/hip_runtime.h>

typedef __attribute__((ext_vector_type(16))) _Float16 v16h;
typedef __attribute__((ext_vector_type(8)))  _Float16 v8h;
typedef __attribute__((ext_vector_type(8)))  float    v8f;

#define NV 32000
#define NE 256
#define NH 512
#define NB 32
#define NS 128
#define NT 64
#define TD (NT - 1)   // 63 decode steps
#define NX (NE + NH)  // 768 decoder LSTM input width

#define RWG 8         // workgroups for persistent recurrent kernels
#define RTHR 256      // threads per recurrent workgroup (8 waves)

#if __has_builtin(__builtin_amdgcn_tensor_load_to_lds) && \
    __has_builtin(__builtin_amdgcn_s_wait_tensorcnt)
#define HAVE_TDM 1
#endif

// ---------------- WMMA helpers (CDNA5 16x16x32 f16 -> f32) ----------------

__device__ __forceinline__ v8f wmma16(v16h a, v16h b, v8f c) {
  return __builtin_amdgcn_wmma_f32_16x16x32_f16(false, a, false, b, (short)0, c,
                                                false, false);
}

// A tile (16x32, MxK), row-major source. Lane L<16: M=L, K=k0+{0..15};
// lane L>=16: M=L-16, K=k0+{16..31}.
__device__ __forceinline__ v16h load_a16(const _Float16* A, int lda, int row0,
                                         int k0, int lane) {
  int m  = lane & 15;
  int kh = (lane >> 4) << 4;
  const _Float16* p = A + (size_t)(row0 + m) * lda + (k0 + kh);
  return *(const v16h*)(p);
}

// B tile (32x16, KxN) with B[k][n] = W[n][k], W row-major.
__device__ __forceinline__ v16h load_b16(const _Float16* W, int ldw, int n0,
                                         int k0, int lane) {
  int n  = lane & 15;
  int ko = (lane >> 4) << 4;
  return *(const v16h*)(W + (size_t)(n0 + n) * ldw + (k0 + ko));
}

__device__ __forceinline__ float sigm(float x) { return 1.f / (1.f + __expf(-x)); }

// Monotonic grid barrier: counter zeroed in-stream before each persistent launch.
__device__ __forceinline__ void grid_barrier(int* cnt, int target) {
  __syncthreads();
  if (threadIdx.x == 0) {
    __hip_atomic_fetch_add(cnt, 1, __ATOMIC_ACQ_REL, __HIP_MEMORY_SCOPE_AGENT);
    while (__hip_atomic_load(cnt, __ATOMIC_ACQUIRE, __HIP_MEMORY_SCOPE_AGENT) <
           target) {
      __builtin_amdgcn_s_sleep(1);
    }
  }
  __syncthreads();
}

// ---------------- prep kernels ----------------

__global__ void k_zero_i32(int* __restrict__ p, int n) {
  int i = blockIdx.x * blockDim.x + threadIdx.x;
  if (i < n) p[i] = 0;
}

__global__ void k_f32_to_f16(const float* __restrict__ in,
                             _Float16* __restrict__ out, int n) {
  int i = blockIdx.x * blockDim.x + threadIdx.x;
  if (i < n) out[i] = (_Float16)in[i];
}

__global__ void k_split_attn(const float* __restrict__ aw,  // (NH, 2*NH)
                             _Float16* __restrict__ Wh, _Float16* __restrict__ We) {
  int i = blockIdx.x * blockDim.x + threadIdx.x;
  if (i < NH * NH) {
    int n = i >> 9, k = i & (NH - 1);
    Wh[i] = (_Float16)aw[n * 2 * NH + k];
    We[i] = (_Float16)aw[n * 2 * NH + NH + k];
  }
}

__global__ void k_gather_src(const int* __restrict__ src,
                             const float* __restrict__ embed,
                             _Float16* __restrict__ out) {  // (NS, NB, NE)
  int i = blockIdx.x * blockDim.x + threadIdx.x;
  if (i < NS * NB * NE) {
    int e = i & (NE - 1);
    int r = i >> 8;
    int b = r & (NB - 1);
    int s = r >> 5;
    out[i] = (_Float16)embed[(size_t)src[b * NS + s] * NE + e];
  }
}

__global__ void k_gather_tgt(const int* __restrict__ tgt,
                             const float* __restrict__ embed,
                             _Float16* __restrict__ out) {  // (TD, NB, NE)
  int i = blockIdx.x * blockDim.x + threadIdx.x;
  if (i < TD * NB * NE) {
    int e = i & (NE - 1);
    int r = i >> 8;
    int b = r & (NB - 1);
    int t = r >> 5;
    out[i] = (_Float16)embed[(size_t)tgt[b * NT + t] * NE + e];
  }
}

// ---------------- encoder: 8-WGP persistent LSTM, grid barrier per step ----------------
// 64 waves total; wave gw owns gate tile task (mt = gw>>5, nt = gw&31) for all 4 gates.
// c stays in VGPRs; h double-buffered in global (L2-resident).

__global__ __launch_bounds__(RTHR, 1) void k_encoder(
    const _Float16* __restrict__ xemb,  // (NS, NB, NE)
    const _Float16* __restrict__ Wih,   // (4H, NE)
    const _Float16* __restrict__ Whh,   // (4H, NH)
    const float* __restrict__ bih, const float* __restrict__ bhh,
    _Float16* __restrict__ enc_out,     // (NS*NB, NH), row = s*NB + b
    _Float16* __restrict__ hG,          // (2, NB, NH) f16 double buffer
    _Float16* __restrict__ hfin,        // (NB, NH)
    float* __restrict__ cfin,           // (NB, NH)
    int* __restrict__ bar) {
  const int tid = threadIdx.x, lane = tid & 31;
  const int gw = blockIdx.x * (RTHR / 32) + (tid >> 5);  // 0..63
  const int mt = gw >> 5, nt = gw & 31;
  const int N = nt * 16 + (lane & 15);
  float creg[8] = {};
  float bg[4];
#pragma unroll
  for (int g = 0; g < 4; ++g) bg[g] = bih[g * NH + N] + bhh[g * NH + N];
  int nb = 0;

  for (int t = 0; t < NS; ++t) {
    const _Float16* X = xemb + (size_t)t * NB * NE;
    v8f acc[4] = {};
    for (int k = 0; k < NE; k += 32) {
      v16h a = load_a16(X, NE, mt * 16, k, lane);
#pragma unroll
      for (int g = 0; g < 4; ++g) {
        v16h b = load_b16(Wih, NE, g * NH + nt * 16, k, lane);
        acc[g] = wmma16(a, b, acc[g]);
      }
    }
    if (t > 0) {  // h == 0 at t==0: skip Whh term entirely
      const _Float16* hc = hG + (t & 1) * NB * NH;
      for (int k = 0; k < NH; k += 32) {
        v16h a = load_a16(hc, NH, mt * 16, k, lane);
#pragma unroll
        for (int g = 0; g < 4; ++g) {
          v16h b = load_b16(Whh, NH, g * NH + nt * 16, k, lane);
          acc[g] = wmma16(a, b, acc[g]);
        }
      }
    }
    _Float16* hn = hG + ((t & 1) ^ 1) * NB * NH;
#pragma unroll
    for (int j = 0; j < 8; ++j) {
      float iv = sigm(acc[0][j] + bg[0]);
      float fv = sigm(acc[1][j] + bg[1]);
      float gv = tanhf(acc[2][j] + bg[2]);
      float ov = sigm(acc[3][j] + bg[3]);
      float c = fv * creg[j] + iv * gv;
      creg[j] = c;
      float hv = ov * tanhf(c);
      int M = mt * 16 + ((lane >> 4) << 3) + j;
      _Float16 h16 = (_Float16)hv;
      hn[M * NH + N] = h16;
      enc_out[((size_t)t * NB + M) * NH + N] = h16;
      if (t == NS - 1) {
        hfin[M * NH + N] = h16;
        cfin[M * NH + N] = c;
      }
    }
    grid_barrier(bar, ++nb * RWG);
  }
}

// ---------------- decoder: 8-WGP persistent attention+LSTM, 3 barriers/step ----------
// Each WG owns 4 batch rows for scores/softmax/context (fused, LDS-local);
// q-projection and LSTM tasks spread over all 64 waves.

__global__ __launch_bounds__(RTHR, 1) void k_decoder(
    const _Float16* __restrict__ xdec,    // (TD, NB, NE)
    const _Float16* __restrict__ Wih,     // (4H, NX)
    const _Float16* __restrict__ Whh,     // (4H, NH)
    const float* __restrict__ bih, const float* __restrict__ bhh,
    const _Float16* __restrict__ Wh,      // (NH, NH)
    const float* __restrict__ attn_v,     // (NH)
    const float* __restrict__ encproj,    // (NS*NB, NH) f32, row = s*NB + b
    const _Float16* __restrict__ enc_out, // (NS*NB, NH) f16
    const _Float16* __restrict__ h0,      // (NB, NH) encoder final h
    const float* __restrict__ c0,         // (NB, NH) encoder final c
    float* __restrict__ qG,               // (NB, NH) f32
    _Float16* __restrict__ xcatG,         // (NB, NX) f16
    _Float16* __restrict__ hG,            // (2, NB, NH) f16 double buffer
    _Float16* __restrict__ hs,            // (NB*TD, NH), row = b*TD + t
    int* __restrict__ bar) {
  __shared__ float sc[4 * NS];   // 4 batch rows x 128 scores
  __shared__ float sav[NH];      // attn_v staged once
  const int tid = threadIdx.x, lane = tid & 31, wv = tid >> 5;
  const int gw = blockIdx.x * (RTHR / 32) + wv;  // 0..63
  const int mt = gw >> 5, nt = gw & 31;
  const int N = nt * 16 + (lane & 15);
  const int brow0 = blockIdx.x * 4;  // this WG's attention batch rows

  for (int i = tid; i < NH; i += RTHR) sav[i] = attn_v[i];
  float creg[8];
  float bg[4];
#pragma unroll
  for (int j = 0; j < 8; ++j) {
    int M = mt * 16 + ((lane >> 4) << 3) + j;
    creg[j] = c0[M * NH + N];
  }
#pragma unroll
  for (int g = 0; g < 4; ++g) bg[g] = bih[g * NH + N] + bhh[g * NH + N];
  int nb = 0;
  __syncthreads();

  for (int t = 0; t < TD; ++t) {
    const _Float16* hc = (t == 0) ? h0 : hG + (t & 1) * NB * NH;

    // Phase A: q = h @ Wh^T (WMMA, one 16x16 tile per wave)
    {
      v8f acc = {};
      for (int k = 0; k < NH; k += 32) {
        v16h a = load_a16(hc, NH, mt * 16, k, lane);
        v16h b = load_b16(Wh, NH, nt * 16, k, lane);
        acc = wmma16(a, b, acc);
      }
#pragma unroll
      for (int j = 0; j < 8; ++j) {
        int M = mt * 16 + ((lane >> 4) << 3) + j;
        qG[M * NH + N] = acc[j];
      }
    }
    grid_barrier(bar, ++nb * RWG);

    // Phase B: scores for this WG's 4 rows (LDS-local)
    for (int idx = tid; idx < 4 * NS; idx += RTHR) {
      int br = idx >> 7, s = idx & (NS - 1);
      int b = brow0 + br;
      const float* ep = encproj + ((size_t)s * NB + b) * NH;
      const float* qb = qG + b * NH;
      float sum = 0.f;
      for (int h = 0; h < NH; ++h) sum += sav[h] * tanhf(qb[h] + ep[h]);
      sc[br * NS + s] = sum;
    }
    __syncthreads();

    // Phase C: softmax, wave w (<4) owns row w (wave32 shfl reduction)
    if (wv < 4) {
      float v[4];
      float mx = -3.4e38f;
#pragma unroll
      for (int i = 0; i < 4; ++i) {
        v[i] = sc[wv * NS + lane + 32 * i];
        mx = fmaxf(mx, v[i]);
      }
#pragma unroll
      for (int off = 16; off > 0; off >>= 1) mx = fmaxf(mx, __shfl_xor(mx, off, 32));
      float s = 0.f;
#pragma unroll
      for (int i = 0; i < 4; ++i) {
        v[i] = __expf(v[i] - mx);
        s += v[i];
      }
#pragma unroll
      for (int off = 16; off > 0; off >>= 1) s += __shfl_xor(s, off, 32);
      float inv = 1.f / s;
#pragma unroll
      for (int i = 0; i < 4; ++i) sc[wv * NS + lane + 32 * i] = v[i] * inv;
    }
    __syncthreads();

    // Phase D: xcat = [x_t | context] for this WG's 4 rows
    for (int idx = tid; idx < 4 * NE; idx += RTHR) {
      int br = idx >> 8, e = idx & (NE - 1);
      int b = brow0 + br;
      xcatG[b * NX + e] = xdec[((size_t)t * NB + b) * NE + e];
    }
    for (int idx = tid; idx < 4 * NH; idx += RTHR) {
      int br = idx >> 9, h = idx & (NH - 1);
      int b = brow0 + br;
      float ctx = 0.f;
      for (int s = 0; s < NS; ++s)
        ctx += sc[br * NS + s] * (float)enc_out[((size_t)s * NB + b) * NH + h];
      xcatG[b * NX + NE + h] = (_Float16)ctx;
    }
    grid_barrier(bar, ++nb * RWG);

    // Phase E: LSTM gates = xcat @ Wih^T + h @ Whh^T (WMMA)
    {
      v8f acc[4] = {};
      for (int k = 0; k < NX; k += 32) {
        v16h a = load_a16(xcatG, NX, mt * 16, k, lane);
#pragma unroll
        for (int g = 0; g < 4; ++g) {
          v16h b = load_b16(Wih, NX, g * NH + nt * 16, k, lane);
          acc[g] = wmma16(a, b, acc[g]);
        }
      }
      for (int k = 0; k < NH; k += 32) {
        v16h a = load_a16(hc, NH, mt * 16, k, lane);
#pragma unroll
        for (int g = 0; g < 4; ++g) {
          v16h b = load_b16(Whh, NH, g * NH + nt * 16, k, lane);
          acc[g] = wmma16(a, b, acc[g]);
        }
      }
      _Float16* hn = hG + ((t & 1) ^ 1) * NB * NH;
#pragma unroll
      for (int j = 0; j < 8; ++j) {
        float iv = sigm(acc[0][j] + bg[0]);
        float fv = sigm(acc[1][j] + bg[1]);
        float gv = tanhf(acc[2][j] + bg[2]);
        float ov = sigm(acc[3][j] + bg[3]);
        float c = fv * creg[j] + iv * gv;
        creg[j] = c;
        float hv = ov * tanhf(c);
        int M = mt * 16 + ((lane >> 4) << 3) + j;
        _Float16 h16 = (_Float16)hv;
        hn[M * NH + N] = h16;
        hs[((size_t)M * TD + t) * NH + N] = h16;
      }
    }
    grid_barrier(bar, ++nb * RWG);
  }
}

// ---------------- parallel GEMM: out[M,N] = A(M,K) @ W(N,K)^T + bias[N] ------------
// Block tile 16(M) x 256(N): A tile staged in LDS (TDM if available), each of 8 waves
// computes two 16x16 C tiles per A fetch. K <= 512.

__global__ __launch_bounds__(256, 1) void k_gemm_bias(
    const _Float16* __restrict__ A, const _Float16* __restrict__ W,
    const float* __restrict__ bias, float* __restrict__ out, int M, int K,
    int lda, int ldw, int ldo) {
  __shared__ _Float16 sA[16 * 512];
  const int wv = threadIdx.x >> 5, lane = threadIdx.x & 31;
  const int m0 = blockIdx.y * 16;
  const int n0 = blockIdx.x * 256 + wv * 32;

#ifdef HAVE_TDM
  typedef __attribute__((ext_vector_type(4))) unsigned int u32x4;
  typedef __attribute__((ext_vector_type(8))) int i32x8;
  typedef __attribute__((ext_vector_type(4))) int i32x4;
  if (wv == 0) {
    // Tensor DMA: 2D tile (16 rows x K f16) from global into LDS offset 0.
    unsigned long long ga = (unsigned long long)(const void*)(A + (size_t)m0 * lda);
    u32x4 g0 = {};
    g0[0] = 1u;                                              // count=1, user mode
    g0[1] = 0u;                                              // lds_addr = 0 (sA)
    g0[2] = (unsigned int)ga;                                // global_addr[31:0]
    g0[3] = (unsigned int)((ga >> 32) & 0x01FFFFFFull) | (2u << 30);  // [56:32]+type=2
    i32x8 g1 = {};
    g1[0] = 1 << 16;                   // data_size = 2 bytes
    g1[1] = (K & 0xFFFF) << 16;        // tensor_dim0 = K          (bits 79:48)
    g1[2] = (M & 0xFFFF) << 16;        // tensor_dim1 = M rows     (bits 111:80)
    g1[3] = (K & 0xFFFF) << 16;        // tile_dim0 = K            (bits 127:112)
    g1[4] = 16;                        // tile_dim1 = 16 rows      (bits 143:128)
    g1[5] = lda;                       // tensor_dim0_stride       (bits 207:160)
    g1[6] = 0;
    g1[7] = 0;
    i32x4 gz = {};
#if __has_include(<hip/amd_detail/amd_gfx1250_TDM.h>)
    i32x8 gz8 = {};
    __builtin_amdgcn_tensor_load_to_lds(g0, g1, gz, gz, gz8, 0);
#else
    __builtin_amdgcn_tensor_load_to_lds(g0, g1, gz, gz, 0);
#endif
    __builtin_amdgcn_s_wait_tensorcnt(0);
  }
#else
  {
    int r = threadIdx.x >> 4;              // 16 threads per row
    int cb = (threadIdx.x & 15) * (K >> 4);
    for (int c = 0; c < (K >> 4); c += 8)
      *(v8h*)(sA + r * K + cb + c) = *(const v8h*)(A + (size_t)(m0 + r) * lda + cb + c);
  }
#endif
  __syncthreads();

  v8f acc0 = {}, acc1 = {};
  for (int k = 0; k < K; k += 32) {
    if (k + 32 < K) {
      __builtin_prefetch(W + (size_t)(n0 + (lane & 15)) * ldw + k + 32, 0, 3);
      __builtin_prefetch(W + (size_t)(n0 + 16 + (lane & 15)) * ldw + k + 32, 0, 3);
    }
    v16h a = load_a16(sA, K, 0, k, lane);
    v16h b0 = load_b16(W, ldw, n0, k, lane);
    v16h b1 = load_b16(W, ldw, n0 + 16, k, lane);
    acc0 = wmma16(a, b0, acc0);
    acc1 = wmma16(a, b1, acc1);
  }
  const int N0 = n0 + (lane & 15);
  const float bz0 = bias[N0], bz1 = bias[N0 + 16];
#pragma unroll
  for (int j = 0; j < 8; ++j) {
    int Mr = m0 + ((lane >> 4) << 3) + j;
    out[(size_t)Mr * ldo + N0] = acc0[j] + bz0;
    out[(size_t)Mr * ldo + N0 + 16] = acc1[j] + bz1;
  }
}

// ---------------- host launch ----------------

extern "C" void kernel_launch(void* const* d_in, const int* in_sizes, int n_in,
                              void* d_out, int out_size, void* d_ws,
                              size_t ws_size, hipStream_t stream) {
  (void)in_sizes; (void)n_in; (void)out_size; (void)ws_size;
  const int*   src     = (const int*)d_in[0];
  const int*   tgt     = (const int*)d_in[1];
  const float* embed   = (const float*)d_in[2];
  const float* enc_Wih = (const float*)d_in[3];
  const float* enc_Whh = (const float*)d_in[4];
  const float* enc_bih = (const float*)d_in[5];
  const float* enc_bhh = (const float*)d_in[6];
  const float* dec_Wih = (const float*)d_in[7];
  const float* dec_Whh = (const float*)d_in[8];
  const float* dec_bih = (const float*)d_in[9];
  const float* dec_bhh = (const float*)d_in[10];
  const float* attn_W  = (const float*)d_in[11];
  const float* attn_b  = (const float*)d_in[12];
  const float* attn_v  = (const float*)d_in[13];
  const float* fc_W    = (const float*)d_in[14];
  const float* fc_b    = (const float*)d_in[15];
  float* out = (float*)d_out;

  char* ws = (char*)d_ws;
  size_t off = 0;
  auto alloc = [&](size_t bytes) {
    char* p = ws + off;
    off = (off + bytes + 255) & ~(size_t)255;
    return p;
  };
  _Float16* eWih16  = (_Float16*)alloc((size_t)4 * NH * NE * 2);
  _Float16* eWhh16  = (_Float16*)alloc((size_t)4 * NH * NH * 2);
  _Float16* dWih16  = (_Float16*)alloc((size_t)4 * NH * NX * 2);
  _Float16* dWhh16  = (_Float16*)alloc((size_t)4 * NH * NH * 2);
  _Float16* fcW16   = (_Float16*)alloc((size_t)NV * NH * 2);
  _Float16* Wh16    = (_Float16*)alloc((size_t)NH * NH * 2);
  _Float16* We16    = (_Float16*)alloc((size_t)NH * NH * 2);
  _Float16* srcemb  = (_Float16*)alloc((size_t)NS * NB * NE * 2);
  _Float16* decin   = (_Float16*)alloc((size_t)TD * NB * NE * 2);
  _Float16* encout  = (_Float16*)alloc((size_t)NS * NB * NH * 2);
  float*    encproj = (float*)alloc((size_t)NS * NB * NH * 4);
  _Float16* h0      = (_Float16*)alloc((size_t)NB * NH * 2);
  float*    c0      = (float*)alloc((size_t)NB * NH * 4);
  _Float16* hsbuf   = (_Float16*)alloc((size_t)NB * TD * NH * 2);
  _Float16* hGenc   = (_Float16*)alloc((size_t)2 * NB * NH * 2);
  _Float16* hGdec   = (_Float16*)alloc((size_t)2 * NB * NH * 2);
  float*    qG      = (float*)alloc((size_t)NB * NH * 4);
  _Float16* xcatG   = (_Float16*)alloc((size_t)NB * NX * 2);
  int*      bars    = (int*)alloc(2 * sizeof(int));

  auto cvt = [&](const float* in, _Float16* o, int n) {
    k_f32_to_f16<<<(n + 255) / 256, 256, 0, stream>>>(in, o, n);
  };
  cvt(enc_Wih, eWih16, 4 * NH * NE);
  cvt(enc_Whh, eWhh16, 4 * NH * NH);
  cvt(dec_Wih, dWih16, 4 * NH * NX);
  cvt(dec_Whh, dWhh16, 4 * NH * NH);
  cvt(fc_W, fcW16, NV * NH);
  k_split_attn<<<(NH * NH + 255) / 256, 256, 0, stream>>>(attn_W, Wh16, We16);
  k_gather_src<<<(NS * NB * NE + 255) / 256, 256, 0, stream>>>(src, embed, srcemb);
  k_gather_tgt<<<(TD * NB * NE + 255) / 256, 256, 0, stream>>>(tgt, embed, decin);
  k_zero_i32<<<1, 64, 0, stream>>>(bars, 2);

  // Encoder recurrence across 8 WGPs with per-step grid barrier
  k_encoder<<<RWG, RTHR, 0, stream>>>(srcemb, eWih16, eWhh16, enc_bih, enc_bhh,
                                      encout, hGenc, h0, c0, bars + 0);

  // enc_proj = enc_out @ W_e^T + attn_b  (4096 x 512, K=512)
  dim3 gp(NH / 256, (NS * NB) / 16);
  k_gemm_bias<<<gp, 256, 0, stream>>>(encout, We16, attn_b, encproj,
                                      NS * NB, NH, NH, NH, NH);

  // Decoder recurrence across 8 WGPs, 3 grid barriers per step
  k_decoder<<<RWG, RTHR, 0, stream>>>(decin, dWih16, dWhh16, dec_bih, dec_bhh,
                                      Wh16, attn_v, encproj, encout, h0, c0, qG,
                                      xcatG, hGdec, hsbuf, bars + 1);

  // Vocab projection: (2016 x 32000, K=512); rows r = b*63+t match (b,t,v) layout
  dim3 go(NV / 256, (NB * TD) / 16);
  k_gemm_bias<<<go, 256, 0, stream>>>(hsbuf, fcW16, fc_b, out, NB * TD, NH, NH,
                                      NH, NV);
}
